// FlashAttentionRope_30468497998146
// MI455X (gfx1250) — matrix operand order repaired and checked by reading, not compiled, not run on hardware
//
#include <hip/hip_runtime.h>
#include <hip/hip_bf16.h>

// ---------------------------------------------------------------------------
// FlashAttention + RoPE + qk-LayerNorm block for MI455X (gfx1250).
// All GEMM-shaped math runs on v_wmma_f32_16x16x32_bf16 (f32 accumulate).
// GEMM tile staging uses GLOBAL_LOAD_ASYNC_TO_LDS_B128 (ASYNCcnt) with LDS
// double-buffering when the toolchain exposes the builtin.
// Shapes fixed by the reference: B=2, N=2048, DIM=1024, H=16, D=64.
// ---------------------------------------------------------------------------

#define Bb   2
#define Nn   2048
#define DIMc 1024
#define Hh   16
#define Dd   64
#define TOK  (Bb * Nn)          // 4096 tokens
#define SCALE 0.125f            // D^-0.5 = 1/8

typedef __attribute__((ext_vector_type(16))) __bf16 v16bf;
typedef __attribute__((ext_vector_type(8)))  __bf16 v8bf;
typedef __attribute__((ext_vector_type(8)))  float  v8f;
typedef __attribute__((ext_vector_type(4)))  int    i32x4;

#if __has_builtin(__builtin_amdgcn_global_load_async_to_lds_b128)
#define HAVE_ASYNC_LDS 1
// param0: i32x4 in AS1 (global), param1: i32x4 in AS3 (LDS)
#define GCAST(p) ((__attribute__((address_space(1))) i32x4*)(p))
#define LCAST(p) ((__attribute__((address_space(3))) i32x4*)(p))
#else
#define HAVE_ASYNC_LDS 0
#endif

static __device__ __forceinline__ void wait_async0() {
#if HAVE_ASYNC_LDS
#if __has_builtin(__builtin_amdgcn_s_wait_asynccnt)
    __builtin_amdgcn_s_wait_asynccnt(0);
#else
    asm volatile("s_wait_asynccnt 0" ::: "memory");
#endif
#endif
}

static __device__ __forceinline__ v16bf join16(v8bf lo, v8bf hi) {
    return __builtin_shufflevector(lo, hi, 0,1,2,3,4,5,6,7,8,9,10,11,12,13,14,15);
}

static __device__ __forceinline__ v8f wmma_bf16(v16bf a, v16bf b, v8f c) {
    // (neg_a, A, neg_b, B, c_mod, C, reuse_a, reuse_b)
    return __builtin_amdgcn_wmma_f32_16x16x32_bf16(false, a, false, b, (short)0, c, false, false);
}

// ---------------------------------------------------------------------------
// fp32 -> bf16 grid-stride converter
// ---------------------------------------------------------------------------
__global__ void cvt_f32_bf16_kernel(const float* __restrict__ in,
                                    __bf16* __restrict__ out, int n) {
    for (int i = blockIdx.x * blockDim.x + threadIdx.x; i < n;
         i += gridDim.x * blockDim.x) {
        out[i] = (__bf16)in[i];
    }
}

// ---------------------------------------------------------------------------
// C[M,N] = A[M,K] * W[N,K]^T (+bias), A/W bf16 row-major, C f32 row-major.
// Block tile 128x64, K-step 32, 256 threads = 8 waves, each wave does a
// 32x32 sub-tile as four 16x16x32 WMMAs. Double-buffered LDS staging via
// async global->LDS copies; padded LDS stride 40 (80B rows) staggers banks.
// Fragment layouts as loaded here (wave32; A and B share one k order):
//   A 16x32 : lane L -> row M=L%16; elems 0..15 = K (L/16)*16..+15
//             of A row M (two 16B LDS loads)
//   B 32x16 : lane L -> col N=L%16; elems = K (L/16)*16..+15 of W row N
//             (one 32B = two 16B LDS loads; W rows are K-contiguous)
//   C 16x16 : lane L -> col N=L%16; VGPR r -> row M = r + 8*(L/16)
// ---------------------------------------------------------------------------
#define GBM 128
#define GBN 64
#define GBK 32
#define GLS 40   // padded LDS row stride (elements)

__global__ __launch_bounds__(256)
void gemm_bf16_wmma_kernel(const __bf16* __restrict__ A,
                           const __bf16* __restrict__ W,
                           const float* __restrict__ bias,
                           float* __restrict__ C,
                           int M, int N, int K) {
    __shared__ __attribute__((aligned(16))) __bf16 As[2 * GBM * GLS];
    __shared__ __attribute__((aligned(16))) __bf16 Ws[2 * GBN * GLS];

    const int tid  = threadIdx.x;
    const int wave = tid >> 5;
    const int lane = tid & 31;
    const int hlf  = lane >> 4;    // which 16-lane half
    const int lr   = lane & 15;
    const int wm   = wave & 3;     // 4 wave-rows
    const int wn   = wave >> 2;    // 2 wave-cols
    const int m0   = blockIdx.y * GBM;
    const int n0   = blockIdx.x * GBN;

    // per-thread staging assignment: A tile = 512 x 16B chunks (2/thread),
    // W tile = 256 x 16B chunks (1/thread)
    const int sRow = tid >> 2;
    const int sCh  = (tid & 3) * 8;
    const __bf16* aSrc0 = A + (size_t)(m0 + sRow) * K + sCh;
    const __bf16* aSrc1 = aSrc0 + (size_t)64 * K;
    const __bf16* wSrc  = W + (size_t)(n0 + sRow) * K + sCh;
    const int aDst0 = sRow * GLS + sCh;
    const int aDst1 = aDst0 + 64 * GLS;
    const int wDst  = sRow * GLS + sCh;

    auto stage = [&](int k0, int buf) {
        __bf16* Ab = As + buf * (GBM * GLS);
        __bf16* Wb = Ws + buf * (GBN * GLS);
#if HAVE_ASYNC_LDS
        __builtin_amdgcn_global_load_async_to_lds_b128(
            GCAST(aSrc0 + k0), LCAST(&Ab[aDst0]), 0, 0);
        __builtin_amdgcn_global_load_async_to_lds_b128(
            GCAST(aSrc1 + k0), LCAST(&Ab[aDst1]), 0, 0);
        __builtin_amdgcn_global_load_async_to_lds_b128(
            GCAST(wSrc + k0), LCAST(&Wb[wDst]), 0, 0);
#else
        *(v8bf*)&Ab[aDst0] = *(const v8bf*)(aSrc0 + k0);
        *(v8bf*)&Ab[aDst1] = *(const v8bf*)(aSrc1 + k0);
        *(v8bf*)&Wb[wDst]  = *(const v8bf*)(wSrc + k0);
#endif
    };

    v8f acc[2][2];
    for (int i = 0; i < 2; ++i)
        for (int j = 0; j < 2; ++j)
            acc[i][j] = (v8f)0.0f;

    const int nk = K / GBK;
    stage(0, 0);
    wait_async0();
    __syncthreads();

    for (int k = 0; k < nk; ++k) {
        const int buf = k & 1;
        if (k + 1 < nk) {
            stage((k + 1) * GBK, buf ^ 1);  // overlaps compute below
            __builtin_prefetch(aSrc0 + (k + 2) * GBK, 0, 1);  // global_prefetch_b8
        }

        const __bf16* Ab = As + buf * (GBM * GLS);
        const __bf16* Wb = Ws + buf * (GBN * GLS);
        v16bf af[2], bf[2];
#pragma unroll
        for (int mi = 0; mi < 2; ++mi) {
            int row = (wm * 32 + mi * 16 + lr) * GLS;
            af[mi] = join16(*(const v8bf*)&Ab[row + hlf * 16],
                            *(const v8bf*)&Ab[row + hlf * 16 + 8]);
        }
#pragma unroll
        for (int ni = 0; ni < 2; ++ni) {
            int row = (wn * 32 + ni * 16 + lr) * GLS;
            bf[ni] = join16(*(const v8bf*)&Wb[row + hlf * 16],
                            *(const v8bf*)&Wb[row + hlf * 16 + 8]);
        }
#pragma unroll
        for (int mi = 0; mi < 2; ++mi)
#pragma unroll
            for (int ni = 0; ni < 2; ++ni)
                acc[mi][ni] = wmma_bf16(af[mi], bf[ni], acc[mi][ni]);

        wait_async0();   // next tile's async copies have landed
        __syncthreads(); // all waves done reading buf before it is reused
    }

#pragma unroll
    for (int mi = 0; mi < 2; ++mi)
#pragma unroll
        for (int ni = 0; ni < 2; ++ni) {
            int n = n0 + wn * 32 + ni * 16 + lr;
            float bv = bias ? bias[n] : 0.0f;
#pragma unroll
            for (int r = 0; r < 8; ++r) {
                int m = m0 + wm * 32 + mi * 16 + r + 8 * hlf;
                C[(size_t)m * N + n] = acc[mi][ni][r] + bv;
            }
        }
}

// ---------------------------------------------------------------------------
// Per-head LayerNorm (q,k) + RoPE (q,k) + V transpose.
// One wave handles one (b,h,n) row of D=64: lane owns d=lane and d=lane+32.
// Outputs: Qb,Kb [B,H,N,D] bf16 (rows K-contiguous for WMMA A/B frags),
//          Vt    [B,H,D,N] bf16 (rows key-contiguous for PV B frags).
// ---------------------------------------------------------------------------
__global__ __launch_bounds__(256)
void lnrope_kernel(const float* __restrict__ qkv,
                   const float* __restrict__ qg, const float* __restrict__ qb_,
                   const float* __restrict__ kg, const float* __restrict__ kb_,
                   const int* __restrict__ xpos,
                   __bf16* __restrict__ Qb, __bf16* __restrict__ Kb,
                   __bf16* __restrict__ Vt) {
    const int row  = blockIdx.x * 8 + (threadIdx.x >> 5);
    const int lane = threadIdx.x & 31;
    if (row >= Bb * Hh * Nn) return;
    const int b  = row / (Hh * Nn);
    const int h  = (row / Nn) % Hh;
    const int n  = row % Nn;
    const int bh = b * Hh + h;

    const float pos  = (float)xpos[b * Nn + n];
    const float invf = __expf(-(float)lane * (9.210340371976184f / 32.0f)); // 10000^(-2i/64)
    const float ang  = pos * invf;
    const float ca   = __cosf(ang), sa = __sinf(ang);

    const size_t base = ((size_t)(b * Nn + n)) * (3 * DIMc) + h * Dd;

    for (int s = 0; s < 2; ++s) {  // s=0: q, s=1: k
        const float* src = qkv + base + s * DIMc;
        float e0 = src[lane], e1 = src[lane + 32];
        float t = e0 + e1;
        for (int m = 16; m >= 1; m >>= 1) t += __shfl_xor(t, m, 32);
        float mean = t * (1.0f / 64.0f);
        float d0 = e0 - mean, d1 = e1 - mean;
        float vs = d0 * d0 + d1 * d1;
        for (int m = 16; m >= 1; m >>= 1) vs += __shfl_xor(vs, m, 32);
        float inv = rsqrtf(vs * (1.0f / 64.0f) + 1e-5f);
        const float* g = s ? kg : qg;
        const float* be = s ? kb_ : qb_;
        float t1 = d0 * inv * g[lane]      + be[lane];       // first half
        float t2 = d1 * inv * g[lane + 32] + be[lane + 32];  // second half
        float o0 = t1 * ca - t2 * sa;
        float o1 = t1 * sa + t2 * ca;
        __bf16* dst = (s ? Kb : Qb) + ((size_t)bh * Nn + n) * Dd;
        dst[lane]      = (__bf16)o0;
        dst[lane + 32] = (__bf16)o1;
    }
    // V: transpose to [B,H,D,N]
    const float* vsrc = qkv + base + 2 * DIMc;
    Vt[((size_t)bh * Dd + lane)      * Nn + n] = (__bf16)vsrc[lane];
    Vt[((size_t)bh * Dd + lane + 32) * Nn + n] = (__bf16)vsrc[lane + 32];
}

// ---------------------------------------------------------------------------
// Flash attention: grid (N/64, B*H), 128 threads = 4 waves; each wave owns a
// 16-query tile, streams 32 keys/iteration.
//   S(16x32)  = Q(16x64) * K^T         -> 4 WMMAs (2 kk x 2 jn)
//   online softmax (16-lane shuffles matching C layout)
//   P relayout C->A via LDS (bf16), O(16x64) += P(16x32) * V(32x64) -> 4 WMMAs
// Output written bf16 as [B,N,H*D] so the final projection consumes it
// directly as a row-major A matrix.
// ---------------------------------------------------------------------------
#define FPS 40  // padded LDS stride for P tiles

__global__ __launch_bounds__(128)
void flash_attn_wmma_kernel(const __bf16* __restrict__ Q,
                            const __bf16* __restrict__ Kk,
                            const __bf16* __restrict__ Vt,
                            __bf16* __restrict__ Obf) {
    __shared__ __attribute__((aligned(16))) __bf16 Plds[4 * 16 * FPS];

    const int wv   = threadIdx.x >> 5;
    const int lane = threadIdx.x & 31;
    const int hlf  = lane >> 4;
    const int lr   = lane & 15;
    const int bh   = blockIdx.y;
    const int b    = bh / Hh;
    const int h    = bh % Hh;
    const int q0   = blockIdx.x * 64 + wv * 16;
    __bf16* myP    = &Plds[wv * 16 * FPS];

    // Q A-fragments (K-dim 64 -> two 16x32 frags), direct from global
    const __bf16* qrow = Q + ((size_t)bh * Nn + q0 + lr) * Dd;
    v16bf aQ[2];
#pragma unroll
    for (int kk = 0; kk < 2; ++kk)
        aQ[kk] = join16(*(const v8bf*)(qrow + kk * 32 + hlf * 16),
                        *(const v8bf*)(qrow + kk * 32 + hlf * 16 + 8));

    float rowm[8], rowl[8];
    v8f accO[4];
#pragma unroll
    for (int r = 0; r < 8; ++r) { rowm[r] = -1e30f; rowl[r] = 0.0f; }
#pragma unroll
    for (int dn = 0; dn < 4; ++dn) accO[dn] = (v8f)0.0f;

    for (int j0 = 0; j0 < Nn; j0 += 32) {
        // ---- S = Q * K^T (scaled) ----
        v8f s[2];
#pragma unroll
        for (int jn = 0; jn < 2; ++jn) {
            const __bf16* krow = Kk + ((size_t)bh * Nn + j0 + jn * 16 + lr) * Dd;
            v8f acc = (v8f)0.0f;
#pragma unroll
            for (int kk = 0; kk < 2; ++kk) {
                v16bf bK = join16(*(const v8bf*)(krow + kk * 32 + hlf * 16),
                                  *(const v8bf*)(krow + kk * 32 + hlf * 16 + 8));
                acc = wmma_bf16(aQ[kk], bK, acc);
            }
            s[jn] = acc;
        }
        // ---- online softmax over the 32 new columns ----
#pragma unroll
        for (int r = 0; r < 8; ++r) {
            float s0 = s[0][r] * SCALE, s1 = s[1][r] * SCALE;
            float tm = fmaxf(s0, s1);
            for (int m = 8; m >= 1; m >>= 1) tm = fmaxf(tm, __shfl_xor(tm, m, 32));
            float nm    = fmaxf(rowm[r], tm);
            float alpha = __expf(rowm[r] - nm);
            float p0    = __expf(s0 - nm);
            float p1    = __expf(s1 - nm);
            float ps    = p0 + p1;
            for (int m = 8; m >= 1; m >>= 1) ps += __shfl_xor(ps, m, 32);
            rowl[r] = rowl[r] * alpha + ps;
            rowm[r] = nm;
#pragma unroll
            for (int dn = 0; dn < 4; ++dn) accO[dn][r] *= alpha;
            int mloc = r + 8 * hlf;  // C layout: VGPR r holds row r + 8*half
            myP[mloc * FPS + lr]      = (__bf16)p0;
            myP[mloc * FPS + 16 + lr] = (__bf16)p1;
        }
        __builtin_amdgcn_wave_barrier();
        asm volatile("s_wait_dscnt 0" ::: "memory");
        __builtin_amdgcn_wave_barrier();

        // ---- O += P * V ----
        v16bf aP = join16(*(const v8bf*)&myP[lr * FPS + hlf * 16],
                          *(const v8bf*)&myP[lr * FPS + hlf * 16 + 8]);
#pragma unroll
        for (int dn = 0; dn < 4; ++dn) {
            const __bf16* vrow = Vt + ((size_t)bh * Dd + dn * 16 + lr) * Nn
                                 + j0 + hlf * 16;
            v16bf bV = join16(*(const v8bf*)(vrow), *(const v8bf*)(vrow + 8));
            accO[dn] = wmma_bf16(aP, bV, accO[dn]);
        }
        __builtin_amdgcn_wave_barrier();  // keep P stores of next iter behind loads
    }

    // ---- normalize + write O as [B, N, H*D] bf16 ----
#pragma unroll
    for (int r = 0; r < 8; ++r) {
        float rl = 1.0f / rowl[r];
        int m = q0 + r + 8 * hlf;
#pragma unroll
        for (int dn = 0; dn < 4; ++dn) {
            int col = h * Dd + dn * 16 + lr;
            Obf[((size_t)b * Nn + m) * DIMc + col] = (__bf16)(accO[dn][r] * rl);
        }
    }
}

// ---------------------------------------------------------------------------
// Launch
// ---------------------------------------------------------------------------
extern "C" void kernel_launch(void* const* d_in, const int* in_sizes, int n_in,
                              void* d_out, int out_size, void* d_ws, size_t ws_size,
                              hipStream_t stream) {
    const float* x      = (const float*)d_in[0];
    const float* w_qkv  = (const float*)d_in[1];
    const float* w_proj = (const float*)d_in[2];
    const float* b_proj = (const float*)d_in[3];
    const float* qn_g   = (const float*)d_in[4];
    const float* qn_b   = (const float*)d_in[5];
    const float* kn_g   = (const float*)d_in[6];
    const float* kn_b   = (const float*)d_in[7];
    const int*   xpos   = (const int*)d_in[8];
    float* out = (float*)d_out;

    char* ws = (char*)d_ws;
    // workspace layout (bytes)
    size_t off = 0;
    __bf16* xb   = (__bf16*)(ws + off); off += (size_t)TOK * DIMc * 2;        // 8 MB
    __bf16* wqb  = (__bf16*)(ws + off); off += (size_t)3 * DIMc * DIMc * 2;   // 6 MB
    __bf16* wpb  = (__bf16*)(ws + off); off += (size_t)DIMc * DIMc * 2;       // 2 MB
    float*  qkvf = (float*)(ws + off);  off += (size_t)TOK * 3 * DIMc * 4;    // 48 MB
    __bf16* qb   = (__bf16*)(ws + off); off += (size_t)Bb * Hh * Nn * Dd * 2; // 8 MB
    __bf16* kb   = (__bf16*)(ws + off); off += (size_t)Bb * Hh * Nn * Dd * 2; // 8 MB
    __bf16* vt   = (__bf16*)(ws + off); off += (size_t)Bb * Hh * Dd * Nn * 2; // 8 MB
    __bf16* ob   = (__bf16*)(ws + off); off += (size_t)TOK * DIMc * 2;        // 8 MB

    // 1) fp32 -> bf16
    cvt_f32_bf16_kernel<<<1024, 256, 0, stream>>>(x, xb, TOK * DIMc);
    cvt_f32_bf16_kernel<<<1024, 256, 0, stream>>>(w_qkv, wqb, 3 * DIMc * DIMc);
    cvt_f32_bf16_kernel<<<512, 256, 0, stream>>>(w_proj, wpb, DIMc * DIMc);

    // 2) qkv = x * w_qkv^T   [4096 x 3072]
    gemm_bf16_wmma_kernel<<<dim3((3 * DIMc) / GBN, TOK / GBM), 256, 0, stream>>>(
        xb, wqb, nullptr, qkvf, TOK, 3 * DIMc, DIMc);

    // 3) per-head LN + RoPE on q,k; V transpose
    lnrope_kernel<<<(Bb * Hh * Nn) / 8, 256, 0, stream>>>(
        qkvf, qn_g, qn_b, kn_g, kn_b, xpos, qb, kb, vt);

    // 4) flash attention -> O bf16 [B,N,H*D]
    flash_attn_wmma_kernel<<<dim3(Nn / 64, Bb * Hh), 128, 0, stream>>>(qb, kb, vt, ob);

    // 5) out = O * w_proj^T + b_proj   [4096 x 1024] fp32
    gemm_bf16_wmma_kernel<<<dim3(DIMc / GBN, TOK / GBM), 256, 0, stream>>>(
        ob, wpb, b_proj, out, TOK, DIMc, DIMc);
}
